// Encoder_53257594470456
// MI455X (gfx1250) — matrix operand order repaired and checked by reading, not compiled, not run on hardware
//
#include <hip/hip_runtime.h>
#include <math.h>
#include <stdint.h>

// ---------------------------------------------------------------------------
// CDNA5 (gfx1250, wave32) encoder layer: bf16 WMMA GEMMs + async-LDS staging
// ---------------------------------------------------------------------------
typedef __bf16 bf16_t;
typedef __attribute__((ext_vector_type(16))) __bf16 v16bf;
typedef __attribute__((ext_vector_type(8)))  __bf16 v8bf;
typedef __attribute__((ext_vector_type(8)))  float  v8f;
typedef __attribute__((ext_vector_type(4)))  int    v4i;

#define BK   32   // K per WMMA stage
#define LDSS 40   // row stride (bf16): 80B rows -> 16B aligned, staggered banks

#if defined(__gfx1250__) && __has_builtin(__builtin_amdgcn_global_load_async_to_lds_b128) && __has_builtin(__builtin_amdgcn_s_wait_asynccnt)
#define HAS_ASYNC 1
#else
#define HAS_ASYNC 0
#endif

typedef v4i __attribute__((address_space(1))) as1_v4i;   // prints as "int4 __device__ *"
typedef v4i __attribute__((address_space(3))) as3_v4i;   // LDS

__device__ __forceinline__ void cp16(const bf16_t* g, bf16_t* l) {
#if HAS_ASYNC
  // GLOBAL_LOAD_ASYNC_TO_LDS_B128: tracked with ASYNCcnt, no VGPR data regs
  __builtin_amdgcn_global_load_async_to_lds_b128(
      (as1_v4i*)(uintptr_t)g, (as3_v4i*)(uint32_t)(uintptr_t)l, 0, 0);
#else
  *(v8bf*)l = *(const v8bf*)g;
#endif
}
__device__ __forceinline__ void async_wait() {
#if HAS_ASYNC
  __builtin_amdgcn_s_wait_asynccnt(0);
#endif
}

__device__ __forceinline__ bf16_t f2bf(float f) {
  union { float f; unsigned u; } a; a.f = f;
  unsigned r = a.u + 0x7FFFu + ((a.u >> 16) & 1u);  // round-to-nearest-even
  unsigned short s = (unsigned short)(r >> 16);
  union { unsigned short s; bf16_t b; } c; c.s = s;
  return c.b;
}

// ---------------------------------------------------------------------------
// Batched GEMM: D[z] = act( scale * (A[z] @ B[z]^T) + bias )
//   A: [M][K]  (bf16, or f32 converted in-stage when AF32)
//   B: [N][K]  bf16 (all weights pre-transposed; K/V used K-contiguous)
//   Outputs: optional f32 C, optional bf16 Cbf (vtrans=1 -> per-head [B,H,64,L])
// 256 threads = 8 waves; block tile (WROWS*TM*16) x (WCOLS*TN*16);
// double-buffered LDS, async global->LDS staging of next K-slab.
// ---------------------------------------------------------------------------
template<int WROWS, int WCOLS, int TM, int TN, bool AF32>
__global__ __launch_bounds__(256) void gemm_tb(
    const void* __restrict__ Aq, const bf16_t* __restrict__ B,
    const float* __restrict__ bias,
    float* __restrict__ C, bf16_t* __restrict__ Cbf,
    int M, int N, int K, int lda, int ldb, int ldc,
    float scale, int act, int vtrans,
    long long sAb, long long sAh, long long sBb, long long sBh,
    long long sCb, long long sCh, int Hdim, int Ltok)
{
  constexpr int BMT = WROWS * TM * 16;
  constexpr int BNT = WCOLS * TN * 16;

  __shared__ __align__(16) bf16_t sA[2][BMT][LDSS];
  __shared__ __align__(16) bf16_t sB[2][BNT][LDSS];

  const int zb = blockIdx.z / Hdim;
  const int zh = blockIdx.z % Hdim;
  const bf16_t* Abf = (const bf16_t*)Aq + zb * sAb + zh * sAh;
  const float*  Af  = (const float*)Aq  + zb * sAb + zh * sAh;
  const bf16_t* Bb  = B + zb * sBb + zh * sBh;

  const int m0   = blockIdx.y * BMT;
  const int n0   = blockIdx.x * BNT;
  const int lane = threadIdx.x & 31;
  const int wave = threadIdx.x >> 5;
  const int wm   = wave / WCOLS;
  const int wn   = wave % WCOLS;

  const v8f vzero = {0.f,0.f,0.f,0.f,0.f,0.f,0.f,0.f};
  v8f acc[TM][TN];
  #pragma unroll
  for (int a = 0; a < TM; ++a)
    #pragma unroll
    for (int b = 0; b < TN; ++b) acc[a][b] = vzero;

  auto stageA = [&](int buf, int kk0) {
    if (AF32) {
      #pragma unroll
      for (int c = threadIdx.x; c < BMT * 8; c += 256) {
        int row  = c >> 3;
        int coff = (c & 7) << 2;
        float4 f = *(const float4*)(Af + (long long)(m0 + row) * lda + kk0 + coff);
        sA[buf][row][coff + 0] = f2bf(f.x);
        sA[buf][row][coff + 1] = f2bf(f.y);
        sA[buf][row][coff + 2] = f2bf(f.z);
        sA[buf][row][coff + 3] = f2bf(f.w);
      }
    } else {
      #pragma unroll
      for (int c = threadIdx.x; c < BMT * 4; c += 256) {
        int row  = c >> 2;
        int koff = (c & 3) << 3;
        cp16(Abf + (long long)(m0 + row) * lda + kk0 + koff, &sA[buf][row][koff]);
      }
    }
  };
  auto stageB = [&](int buf, int kk0) {
    #pragma unroll
    for (int c = threadIdx.x; c < BNT * 4; c += 256) {
      int nn   = c >> 2;
      int koff = (c & 3) << 3;
      cp16(Bb + (long long)(n0 + nn) * ldb + kk0 + koff, &sB[buf][nn][koff]);
    }
  };

  const int nsteps = K / BK;
  stageA(0, 0);
  stageB(0, 0);

  const int rsel = lane & 15;
  const int ac0  = (lane >> 4) << 3;   // A frag: lanes 0-15 K{0-7,16-23}; 16-31 K{8-15,24-31}
  const int bkb  = (lane >> 4) << 3;   // B frag: lanes 0-15 K{0-7,16-23}; 16-31 K{8-15,24-31}

  for (int ks = 0; ks < nsteps; ++ks) {
    async_wait();        // my async stores into 'cur' are complete
    __syncthreads();     // everyone's stores visible; prior reads of 'nxt' done
    const int cur = ks & 1;
    const int nxt = cur ^ 1;
    if (ks + 1 < nsteps) {           // prefetch next K-slab while computing
      stageA(nxt, (ks + 1) * BK);
      stageB(nxt, (ks + 1) * BK);
    }

    v16bf af[TM], bfr[TN];
    #pragma unroll
    for (int tm = 0; tm < TM; ++tm) {
      int r = wm * (TM * 16) + tm * 16 + rsel;
      v8bf lo = *(const v8bf*)&sA[cur][r][ac0];
      v8bf hi = *(const v8bf*)&sA[cur][r][ac0 + 16];
      af[tm] = __builtin_shufflevector(lo, hi,
                 0,1,2,3,4,5,6,7,8,9,10,11,12,13,14,15);
    }
    #pragma unroll
    for (int tn = 0; tn < TN; ++tn) {
      int nn = wn * (TN * 16) + tn * 16 + rsel;
      v8bf lo = *(const v8bf*)&sB[cur][nn][bkb];
      v8bf hi = *(const v8bf*)&sB[cur][nn][bkb + 16];
      bfr[tn] = __builtin_shufflevector(lo, hi,
                 0,1,2,3,4,5,6,7,8,9,10,11,12,13,14,15);
    }
    #pragma unroll
    for (int tm = 0; tm < TM; ++tm)
      #pragma unroll
      for (int tn = 0; tn < TN; ++tn)
        acc[tm][tn] = __builtin_amdgcn_wmma_f32_16x16x32_bf16(
            false, af[tm], false, bfr[tn], (short)0, acc[tm][tn], false, false);
  }

  // --- epilogue (C layout: lanes 0-15 rows 0-7, lanes 16-31 rows 8-15) ---
  float* Cb   = C   ? C   + zb * sCb + zh * sCh : nullptr;
  bf16_t* Cfb = Cbf;   // vtrans computes its own offsets; normal mode uses strides
  if (Cbf && !vtrans) Cfb = Cbf + zb * sCb + zh * sCh;

  const int rlo   = lane & 15;
  const int mhalf = (lane >> 4) << 3;
  #pragma unroll
  for (int tm = 0; tm < TM; ++tm) {
    #pragma unroll
    for (int tn = 0; tn < TN; ++tn) {
      const int n = n0 + wn * (TN * 16) + tn * 16 + rlo;
      const float bb = bias ? bias[n] : 0.0f;
      const int mbase = m0 + wm * (TM * 16) + tm * 16 + mhalf;
      float vals[8];
      #pragma unroll
      for (int i = 0; i < 8; ++i) {
        float v = acc[tm][tn][i] * scale + bb;
        if (act == 1) v = 0.5f * v * (1.0f + erff(v * 0.70710678118654752f));
        vals[i] = v;
      }
      if (Cb) {
        #pragma unroll
        for (int i = 0; i < 8; ++i)
          Cb[(long long)(mbase + i) * ldc + n] = vals[i];
      }
      if (Cfb) {
        if (!vtrans) {
          #pragma unroll
          for (int i = 0; i < 8; ++i)
            Cfb[(long long)(mbase + i) * ldc + n] = f2bf(vals[i]);
        } else {
          // per-head transposed: dst[(b*H + n/64)*64 + n%64][l], l = m % Ltok
          const int bidx = mbase / Ltok;
          const int l    = mbase % Ltok;
          long long row  = ((long long)bidx * Hdim + (n >> 6)) * 64 + (n & 63);
          v8bf pk;
          #pragma unroll
          for (int i = 0; i < 8; ++i) pk[i] = f2bf(vals[i]);
          *(v8bf*)&Cfb[row * Ltok + l] = pk;   // 8 contiguous l's, 16B store
        }
      }
    }
  }
}

// ---------------------------------------------------------------------------
// f32 -> bf16 bulk convert (n % 1024 == 0)
// ---------------------------------------------------------------------------
__global__ __launch_bounds__(256) void conv_bf16(const float* __restrict__ x,
                                                 bf16_t* __restrict__ y)
{
  long long i = ((long long)blockIdx.x * 256 + threadIdx.x) * 4;
  float4 f = *(const float4*)(x + i);
  y[i + 0] = f2bf(f.x); y[i + 1] = f2bf(f.y);
  y[i + 2] = f2bf(f.z); y[i + 3] = f2bf(f.w);
}

// ---------------------------------------------------------------------------
// Weight transpose+convert: src f32 [K][N] -> dst bf16 [N][K]
// ---------------------------------------------------------------------------
__global__ __launch_bounds__(256) void trans_conv(const float* __restrict__ src,
                                                  bf16_t* __restrict__ dst,
                                                  int K, int N)
{
  __shared__ float t[32][33];
  const int n0 = blockIdx.x * 32, k0 = blockIdx.y * 32;
  const int tx = threadIdx.x, ty = threadIdx.y;   // 32 x 8
  #pragma unroll
  for (int i = 0; i < 4; ++i)
    t[ty + i * 8][tx] = src[(long long)(k0 + ty + i * 8) * N + n0 + tx];
  __syncthreads();
  #pragma unroll
  for (int i = 0; i < 4; ++i)
    dst[(long long)(n0 + ty + i * 8) * K + k0 + tx] = f2bf(t[tx][ty + i * 8]);
}

// ---------------------------------------------------------------------------
// In-place row softmax (cols == 2048)
// ---------------------------------------------------------------------------
__global__ __launch_bounds__(256) void softmax_inplace(float* __restrict__ x, int cols)
{
  __shared__ float red[256];
  const int t = threadIdx.x;
  float* row = x + (long long)blockIdx.x * cols;

  float v[8];
  float mx = -3.402823466e38f;
  #pragma unroll
  for (int i = 0; i < 8; ++i) { v[i] = row[t + (i << 8)]; mx = fmaxf(mx, v[i]); }
  red[t] = mx; __syncthreads();
  for (int s = 128; s > 0; s >>= 1) { if (t < s) red[t] = fmaxf(red[t], red[t + s]); __syncthreads(); }
  mx = red[0]; __syncthreads();

  float sum = 0.f;
  #pragma unroll
  for (int i = 0; i < 8; ++i) { v[i] = __expf(v[i] - mx); sum += v[i]; }
  red[t] = sum; __syncthreads();
  for (int s = 128; s > 0; s >>= 1) { if (t < s) red[t] += red[t + s]; __syncthreads(); }
  sum = red[0]; __syncthreads();

  const float inv = 1.0f / sum;
  #pragma unroll
  for (int i = 0; i < 8; ++i) row[t + (i << 8)] = v[i] * inv;
}

// ---------------------------------------------------------------------------
// out = LayerNorm(x + r)*g + b ; optional bf16 mirror for downstream GEMMs
// ---------------------------------------------------------------------------
__global__ __launch_bounds__(256) void layernorm_residual(
    const float* __restrict__ x, const float* __restrict__ r,
    const float* __restrict__ g, const float* __restrict__ b,
    float* __restrict__ out, bf16_t* __restrict__ outbf, int D)
{
  __shared__ float red[256];
  const int t = threadIdx.x;
  const long long base = (long long)blockIdx.x * D;

  float v[4];
  float s = 0.f;
  #pragma unroll
  for (int i = 0; i < 4; ++i) {
    int c = t + (i << 8);
    v[i] = x[base + c] + r[base + c];
    s += v[i];
  }
  red[t] = s; __syncthreads();
  for (int st = 128; st > 0; st >>= 1) { if (t < st) red[t] += red[t + st]; __syncthreads(); }
  const float mean = red[0] / (float)D; __syncthreads();

  float vs = 0.f;
  #pragma unroll
  for (int i = 0; i < 4; ++i) { float d = v[i] - mean; vs += d * d; }
  red[t] = vs; __syncthreads();
  for (int st = 128; st > 0; st >>= 1) { if (t < st) red[t] += red[t + st]; __syncthreads(); }
  const float rstd = rsqrtf(red[0] / (float)D + 1e-6f); __syncthreads();

  #pragma unroll
  for (int i = 0; i < 4; ++i) {
    int c = t + (i << 8);
    float o = (v[i] - mean) * rstd * g[c] + b[c];
    if (out)   out[base + c] = o;
    if (outbf) outbf[base + c] = f2bf(o);
  }
}

// ---------------------------------------------------------------------------
// Launch: full encoder layer
// ---------------------------------------------------------------------------
extern "C" void kernel_launch(void* const* d_in, const int* in_sizes, int n_in,
                              void* d_out, int out_size, void* d_ws, size_t ws_size,
                              hipStream_t stream)
{
  const float* rels = (const float*)d_in[0];
  const float* his  = (const float*)d_in[1];
  const float* Wq   = (const float*)d_in[2];
  const float* bq   = (const float*)d_in[3];
  const float* Wk   = (const float*)d_in[4];
  const float* bk   = (const float*)d_in[5];
  const float* Wv   = (const float*)d_in[6];
  const float* bv   = (const float*)d_in[7];
  const float* Wo   = (const float*)d_in[8];
  const float* ln1g = (const float*)d_in[9];
  const float* ln1b = (const float*)d_in[10];
  const float* W1   = (const float*)d_in[11];
  const float* b1   = (const float*)d_in[12];
  const float* W2   = (const float*)d_in[13];
  const float* b2   = (const float*)d_in[14];
  const float* ln2g = (const float*)d_in[15];
  const float* ln2b = (const float*)d_in[16];

  constexpr int Bn = 2, L = 2048, EMBc = 1024, Hn = 16, DK = 64, DFFc = 4096;
  constexpr int BL = Bn * L;                        // 4096 tokens
  constexpr long long TOK = (long long)BL * EMBc;   // 4,194,304
  constexpr long long WSQ = (long long)EMBc * EMBc; // 1,048,576

  // ---- workspace carve (bf16 region first, then f32 region; ~167 MB) ----
  bf16_t* wb      = (bf16_t*)d_ws;
  bf16_t* rels_bf = wb; wb += TOK;
  bf16_t* his_bf  = wb; wb += TOK;
  bf16_t* q_bf    = wb; wb += TOK;
  bf16_t* k_bf    = wb; wb += TOK;
  bf16_t* vT      = wb; wb += TOK;     // [B,H,64,L]
  bf16_t* ctx_bf  = wb; wb += TOK;
  bf16_t* x1_bf   = wb; wb += TOK;
  bf16_t* WqT     = wb; wb += WSQ;
  bf16_t* WkT     = wb; wb += WSQ;
  bf16_t* WvT     = wb; wb += WSQ;
  bf16_t* WoT     = wb; wb += WSQ;
  bf16_t* W1T     = wb; wb += TOK;     // [4096][1024]
  bf16_t* W2T     = wb; wb += TOK;     // [1024][4096]
  bf16_t* h_bf    = wb; wb += 4 * TOK; // [4096][4096]
  float*  y1      = (float*)wb;
  float*  x1      = y1 + TOK;
  float*  ff      = x1 + TOK;

  float* outp = (float*)d_out;
  float* attn = outp + TOK;            // [B,H,L,L]

  const long long LLL = (long long)L * L;
  const long long LE  = (long long)L * EMBc;

  // ---- pre-pass: convert activations, transpose+convert weights ----
  conv_bf16<<<dim3(TOK / 1024), dim3(256), 0, stream>>>(rels, rels_bf);
  conv_bf16<<<dim3(TOK / 1024), dim3(256), 0, stream>>>(his,  his_bf);
  trans_conv<<<dim3(EMBc/32, EMBc/32), dim3(32, 8), 0, stream>>>(Wq, WqT, EMBc, EMBc);
  trans_conv<<<dim3(EMBc/32, EMBc/32), dim3(32, 8), 0, stream>>>(Wk, WkT, EMBc, EMBc);
  trans_conv<<<dim3(EMBc/32, EMBc/32), dim3(32, 8), 0, stream>>>(Wv, WvT, EMBc, EMBc);
  trans_conv<<<dim3(EMBc/32, EMBc/32), dim3(32, 8), 0, stream>>>(Wo, WoT, EMBc, EMBc);
  trans_conv<<<dim3(DFFc/32, EMBc/32), dim3(32, 8), 0, stream>>>(W1, W1T, EMBc, DFFc);
  trans_conv<<<dim3(EMBc/32, DFFc/32), dim3(32, 8), 0, stream>>>(W2, W2T, DFFc, EMBc);

  // GemmMain: 128x128 block (waves 4x2, each 32x64). GemmCtx: 128x64, A is f32.
  #define GemmMain gemm_tb<4,2,2,4,false>
  #define GemmCtx  gemm_tb<8,1,1,4,true>
  const dim3 blk(256);

  // ---- Q/K projections -> bf16 [BL,1024]; V -> per-head transposed bf16 ----
  GemmMain<<<dim3(EMBc/128, BL/128, 1), blk, 0, stream>>>(
      rels_bf, WqT, bq, nullptr, q_bf, BL, EMBc, EMBc, EMBc, EMBc, EMBc,
      1.0f, 0, 0, 0,0,0,0,0,0, 1, L);
  GemmMain<<<dim3(EMBc/128, BL/128, 1), blk, 0, stream>>>(
      his_bf, WkT, bk, nullptr, k_bf, BL, EMBc, EMBc, EMBc, EMBc, EMBc,
      1.0f, 0, 0, 0,0,0,0,0,0, 1, L);
  GemmMain<<<dim3(EMBc/128, BL/128, 1), blk, 0, stream>>>(
      his_bf, WvT, bv, nullptr, vT, BL, EMBc, EMBc, EMBc, EMBc, EMBc,
      1.0f, 0, /*vtrans=*/1, 0,0,0,0,0,0, Hn, L);

  // ---- scores = (q @ k^T)/8 -> f32 attn slot of d_out (batched 32 heads) ----
  GemmMain<<<dim3(L/128, L/128, Bn*Hn), blk, 0, stream>>>(
      q_bf, k_bf, nullptr, attn, nullptr, L, L, DK, EMBc, EMBc, L,
      0.125f, 0, 0,
      LE, 64,  LE, 64,  (long long)Hn*LLL, LLL,  Hn, L);

  // ---- softmax rows in place ----
  softmax_inplace<<<dim3(Bn*Hn*L), dim3(256), 0, stream>>>(attn, L);

  // ---- ctx = attn @ v : A = f32 attn (converted in-stage), B = vT bf16 ----
  GemmCtx<<<dim3(DK/64, L/128, Bn*Hn), blk, 0, stream>>>(
      attn, vT, nullptr, nullptr, ctx_bf, L, DK, L, L, L, EMBc,
      1.0f, 0, 0,
      (long long)Hn*LLL, LLL,
      (long long)Hn*DK*L, (long long)DK*L,
      LE, 64,  Hn, L);

  // ---- out-proj: ctx @ Wo -> y1 (f32) ----
  GemmMain<<<dim3(EMBc/128, BL/128, 1), blk, 0, stream>>>(
      ctx_bf, WoT, nullptr, y1, nullptr, BL, EMBc, EMBc, EMBc, EMBc, EMBc,
      1.0f, 0, 0, 0,0,0,0,0,0, 1, L);

  // ---- x1 = LN1(rels + y1), plus bf16 mirror for FFN ----
  layernorm_residual<<<dim3(BL), dim3(256), 0, stream>>>(
      rels, y1, ln1g, ln1b, x1, x1_bf, EMBc);

  // ---- h = gelu(x1 @ W1 + b1) -> bf16 ----
  GemmMain<<<dim3(DFFc/128, BL/128, 1), blk, 0, stream>>>(
      x1_bf, W1T, b1, nullptr, h_bf, BL, DFFc, EMBc, EMBc, EMBc, DFFc,
      1.0f, /*gelu*/1, 0, 0,0,0,0,0,0, 1, L);

  // ---- ff = h @ W2 + b2 -> f32 ----
  GemmMain<<<dim3(EMBc/128, BL/128, 1), blk, 0, stream>>>(
      h_bf, W2T, b2, ff, nullptr, BL, EMBc, DFFc, DFFc, DFFc, EMBc,
      1.0f, 0, 0, 0,0,0,0,0,0, 1, L);

  // ---- out = LN2(x1 + ff) -> d_out ----
  layernorm_residual<<<dim3(BL), dim3(256), 0, stream>>>(
      x1, ff, ln2g, ln2b, outp, nullptr, EMBc);

  #undef GemmMain
  #undef GemmCtx
}
